// HLG_51376398795558
// MI455X (gfx1250) — compile-verified, matrix operand order checked by reading, not run on hardware
//
#include <hip/hip_runtime.h>

// ---------------- types for WMMA ----------------
typedef __bf16 v16bf __attribute__((ext_vector_type(16)));
typedef float  v8f   __attribute__((ext_vector_type(8)));
typedef unsigned int u32x4 __attribute__((ext_vector_type(4)));

union FragU { v16bf v; u32x4 q[2]; };

__device__ __forceinline__ unsigned short f2bf(float f) {
  unsigned int u = __float_as_uint(f);
  u += 0x7FFFu + ((u >> 16) & 1u);   // round-to-nearest-even
  return (unsigned short)(u >> 16);
}
__device__ __forceinline__ float bf2f(unsigned short h) {
  return __uint_as_float(((unsigned int)h) << 16);
}

// ---------------- WMMA GEMM: Y[N,128] = act( [X0 | X1][N, K0+K1] @ W[K,128] + b ) ----------------
// Optional per-row gather on X0/X1 (g0/g1). Split-bf16 (hi+lo) => ~f32 accuracy, f32 accumulate.
__global__ __launch_bounds__(256) void hlg_gemm128(
    const float* __restrict__ X0, const float* __restrict__ X1,
    const int* __restrict__ g0, const int* __restrict__ g1,
    const float* __restrict__ W, const float* __restrict__ bias,
    float* __restrict__ Y, long long N, int K0, int K1, int doRelu)
{
  __shared__ __align__(16) unsigned short Ahi[128 * 32];
  __shared__ __align__(16) unsigned short Alo[128 * 32];
  __shared__ __align__(16) unsigned short Bhi[128 * 32];   // stored transposed: [n][k]
  __shared__ __align__(16) unsigned short Blo[128 * 32];

  const int K    = K0 + K1;
  const int tid  = threadIdx.x;
  const int wave = tid >> 5;
  const int lane = tid & 31;
  const int m    = lane & 15;
  const int hlf  = lane >> 4;
  const long long r0 = (long long)blockIdx.x * 128;

  v8f acc[8];
#pragma unroll
  for (int n = 0; n < 8; ++n)
#pragma unroll
    for (int e = 0; e < 8; ++e) acc[n][e] = 0.0f;

  for (int kb = 0; kb < K; kb += 32) {
    __syncthreads();
    // stage A block: 128 rows x 32 k (f32 -> hi/lo bf16)
#pragma unroll
    for (int i = 0; i < 16; ++i) {
      int el  = tid + i * 256;
      int row = el >> 5;
      int k   = el & 31;
      long long rg = r0 + row;
      float v = 0.0f;
      if (rg < N) {
        int gk = kb + k;
        if (gk < K0) {
          long long sr = g0 ? (long long)g0[rg] : rg;
          v = X0[sr * (long long)K0 + gk];
        } else {
          long long sr = g1 ? (long long)g1[rg] : rg;
          v = X1[sr * (long long)K1 + (gk - K0)];
        }
      }
      unsigned short hi = f2bf(v);
      unsigned short lo = f2bf(v - bf2f(hi));
      Ahi[row * 32 + k] = hi;
      Alo[row * 32 + k] = lo;
    }
    // stage W block transposed: [n][k]
#pragma unroll
    for (int i = 0; i < 16; ++i) {
      int el = tid + i * 256;
      int k  = el >> 7;
      int n  = el & 127;
      float v = W[(long long)(kb + k) * 128 + n];
      unsigned short hi = f2bf(v);
      unsigned short lo = f2bf(v - bf2f(hi));
      Bhi[n * 32 + k] = hi;
      Blo[n * 32 + k] = lo;
    }
    __syncthreads();

    // A fragment (16-bit A layout: lanes 0-15 M=0..15 / lanes 16-31 select K sub-halves)
    FragU ahi, alo;
    const int arow = wave * 16 + m;
    ahi.q[0] = *(const u32x4*)&Ahi[arow * 32 + hlf * 8];
    ahi.q[1] = *(const u32x4*)&Ahi[arow * 32 + 16 + hlf * 8];
    alo.q[0] = *(const u32x4*)&Alo[arow * 32 + hlf * 8];
    alo.q[1] = *(const u32x4*)&Alo[arow * 32 + 16 + hlf * 8];

#pragma unroll
    for (int nt = 0; nt < 8; ++nt) {
      FragU bhi, blo;
      const int brow = nt * 16 + m;
      bhi.q[0] = *(const u32x4*)&Bhi[brow * 32 + hlf * 8];
      bhi.q[1] = *(const u32x4*)&Bhi[brow * 32 + 16 + hlf * 8];
      blo.q[0] = *(const u32x4*)&Blo[brow * 32 + hlf * 8];
      blo.q[1] = *(const u32x4*)&Blo[brow * 32 + 16 + hlf * 8];
      acc[nt] = __builtin_amdgcn_wmma_f32_16x16x32_bf16(false, ahi.v, false, bhi.v, (short)0, acc[nt], false, false);
      acc[nt] = __builtin_amdgcn_wmma_f32_16x16x32_bf16(false, ahi.v, false, blo.v, (short)0, acc[nt], false, false);
      acc[nt] = __builtin_amdgcn_wmma_f32_16x16x32_bf16(false, alo.v, false, bhi.v, (short)0, acc[nt], false, false);
    }
  }

  // epilogue: C/D layout — lanes 0-15: N=lane, M=vgpr; lanes 16-31: N=lane-16, M=8+vgpr
#pragma unroll
  for (int nt = 0; nt < 8; ++nt) {
    int col = nt * 16 + m;
    float bv = bias ? bias[col] : 0.0f;
#pragma unroll
    for (int r = 0; r < 8; ++r) {
      long long grow = r0 + wave * 16 + (hlf ? 8 + r : r);
      if (grow < N) {
        float v = acc[nt][r] + bv;
        if (doRelu) v = fmaxf(v, 0.0f);
        Y[grow * 128 + col] = v;
      }
    }
  }
}

// ---------------- elementwise / graph kernels ----------------
__global__ void hlg_zero(float* p, long long n) {
  long long t = (long long)blockIdx.x * blockDim.x + threadIdx.x;
  if (t < n) p[t] = 0.0f;
}

__global__ void hlg_embed_atom(const int* __restrict__ xa, const float* __restrict__ emb,
                               float* __restrict__ y, long long n) {
  long long t = (long long)blockIdx.x * blockDim.x + threadIdx.x;
  if (t >= n * 128) return;
  long long row = t >> 7; int c = t & 127;
  float s = 0.0f;
#pragma unroll
  for (int f = 0; f < 9; ++f) {
    int tok = xa[row * 9 + f];
    s += emb[((long long)f * 64 + tok) * 128 + c];
  }
  y[t] = s;
}

__global__ void hlg_embed_bond(const int* __restrict__ ea, const float* __restrict__ emb,
                               float* __restrict__ y, long long n) {
  long long t = (long long)blockIdx.x * blockDim.x + threadIdx.x;
  if (t >= n * 128) return;
  long long row = t >> 7; int c = t & 127;
  float s = 0.0f;
#pragma unroll
  for (int f = 0; f < 3; ++f) {
    int tok = ea[row * 3 + f];
    s += emb[((long long)f * 16 + tok) * 128 + c];
  }
  y[t] = s;
}

// dst[seg(idx[e])] += src[srcg?srcg[e]:e]   (128-wide rows)
__global__ void hlg_scatter_add(const float* __restrict__ src, const int* __restrict__ srcg,
                                const int* __restrict__ idx, const int* __restrict__ idxg,
                                float* __restrict__ dst, long long nsrc) {
  long long t = (long long)blockIdx.x * blockDim.x + threadIdx.x;
  if (t >= nsrc * 128) return;
  long long e = t >> 7; int c = t & 127;
  long long sr = srcg ? (long long)srcg[e] : e;
  int d = idx[e]; if (idxg) d = idxg[d];
  atomicAdd(&dst[(long long)d * 128 + c], src[sr * 128 + c]);
}

__global__ void hlg_count(const int* __restrict__ idx, const int* __restrict__ idxg,
                          float* __restrict__ cnt, long long n) {
  long long e = (long long)blockIdx.x * blockDim.x + threadIdx.x;
  if (e >= n) return;
  int d = idx[e]; if (idxg) d = idxg[d];
  atomicAdd(&cnt[d], 1.0f);
}

__global__ void hlg_divide(float* __restrict__ buf, const float* __restrict__ cnt, long long nrows) {
  long long t = (long long)blockIdx.x * blockDim.x + threadIdx.x;
  if (t >= nrows * 128) return;
  buf[t] = buf[t] / fmaxf(cnt[t >> 7], 1.0f);
}

__global__ void hlg_avg_gather(const float* __restrict__ x, const int* __restrict__ r,
                               const int* __restrict__ c, float* __restrict__ y, long long nE) {
  long long t = (long long)blockIdx.x * blockDim.x + threadIdx.x;
  if (t >= nE * 128) return;
  long long e = t >> 7; int ch = t & 127;
  y[t] = 0.5f * (x[(long long)r[e] * 128 + ch] + x[(long long)c[e] * 128 + ch]);
}

// per-channel sum & sumsq of v = a (+ b) over n rows -> st[0..127]=sum, st[128..255]=sumsq
__global__ __launch_bounds__(128) void hlg_bn_stats(const float* __restrict__ a, const float* __restrict__ b,
                                                    float* __restrict__ st, long long n) {
  int c = threadIdx.x;
  long long r0 = (long long)blockIdx.x * 256;
  float s = 0.0f, ss = 0.0f;
  for (int i = 0; i < 256; ++i) {
    long long r = r0 + i;
    if (r >= n) break;
    float v = a[r * 128 + c];
    if (b) v += b[r * 128 + c];
    s += v; ss += v * v;
  }
  atomicAdd(&st[c], s);
  atomicAdd(&st[128 + c], ss);
}

// y = relu( ((a(+b)) - mean) * rsqrt(var + eps) ),  biased var (torch BN training-mode)
__global__ void hlg_bn_apply(const float* __restrict__ a, const float* __restrict__ b,
                             const float* __restrict__ st, float* __restrict__ y, long long n) {
  long long t = (long long)blockIdx.x * blockDim.x + threadIdx.x;
  if (t >= n * 128) return;
  int c = t & 127;
  float inv_n = 1.0f / (float)n;
  float mean = st[c] * inv_n;
  float var  = fmaxf(st[128 + c] * inv_n - mean * mean, 0.0f);
  float v = a[t];
  if (b) v += b[t];
  y[t] = fmaxf((v - mean) * rsqrtf(var + 1e-5f), 0.0f);
}

__global__ void hlg_final(const float* __restrict__ pA, const float* __restrict__ pE,
                          const float* __restrict__ pF, const float* __restrict__ w,
                          const float* __restrict__ b0, float* __restrict__ out) {
  int bgi = blockIdx.x * blockDim.x + threadIdx.x;
  if (bgi >= 1024) return;
  float s = 0.0f;
#pragma unroll 4
  for (int c = 0; c < 128; ++c)
    s += (pA[(long long)bgi * 128 + c] + pE[(long long)bgi * 128 + c] + pF[(long long)bgi * 128 + c]) * w[c];
  out[bgi] = s + b0[0];
}

// ---------------- host orchestration ----------------
extern "C" void kernel_launch(void* const* d_in, const int* in_sizes, int n_in,
                              void* d_out, int out_size, void* d_ws, size_t ws_size,
                              hipStream_t stream) {
  (void)in_sizes; (void)n_in; (void)out_size;
  const long long A = 150000, E = 300000, F = 50000, NAS = 250000, FE = 100000, NB = 1024;

  const float* fragments  = (const float*)d_in[0];
  const float* atom_emb   = (const float*)d_in[1];
  const float* bond_emb   = (const float*)d_in[2];
  const float* frag_W     = (const float*)d_in[3];
  const float* frag_b     = (const float*)d_in[4];
  const float* a2a_Wb     = (const float*)d_in[5];
  const float* a2a_bb     = (const float*)d_in[6];
  const float* a2a_Wa     = (const float*)d_in[7];
  const float* a2a_ba     = (const float*)d_in[8];
  const float* a2e_Wa     = (const float*)d_in[9];
  const float* a2e_ba     = (const float*)d_in[10];
  const float* a2f_Wa     = (const float*)d_in[11];
  const float* a2f_ba     = (const float*)d_in[12];
  const float* f2a_Wa     = (const float*)d_in[13];
  const float* f2a_ba     = (const float*)d_in[14];
  const float* f2f_Wa     = (const float*)d_in[15];
  const float* f2f_ba     = (const float*)d_in[16];
  const float* cA_W       = (const float*)d_in[17];
  const float* cA_b       = (const float*)d_in[18];
  const float* cE_W       = (const float*)d_in[19];
  const float* cE_b       = (const float*)d_in[20];
  const float* cF_W       = (const float*)d_in[21];
  const float* cF_b       = (const float*)d_in[22];
  const float* atom_out_W = (const float*)d_in[23];
  const float* atom_out_b = (const float*)d_in[24];
  const float* edge_out_W = (const float*)d_in[25];
  const float* edge_out_b = (const float*)d_in[26];
  const float* frag_out_W = (const float*)d_in[27];
  const float* frag_out_b = (const float*)d_in[28];
  // d_in[29], d_in[30]: mol_out_* — the x_mol==0 path provably contributes exactly 0 (BN of identical rows)
  const float* out_W      = (const float*)d_in[31];
  const float* out_b      = (const float*)d_in[32];
  const int* x_atom    = (const int*)d_in[33];
  const int* edge_attr = (const int*)d_in[34];
  const int* edge_idx  = (const int*)d_in[35];
  const int* batch     = (const int*)d_in[36];
  const int* fai       = (const int*)d_in[37];   // frag_atom_idx
  const int* ffi       = (const int*)d_in[38];   // frag_frag_idx
  const int* fedge     = (const int*)d_in[39];
  const int* frag_batch = (const int*)d_in[40];
  const int* row_e = edge_idx;
  const int* col_e = edge_idx + E;
  const int* fr_row = fedge;
  const int* fr_col = fedge + FE;

  // workspace layout
  float* S0  = (float*)d_ws;
  float* S1  = S0 + E * 128;
  float* S2  = S1 + E * 128;
  float* S3  = S2 + E * 128;
  float* x   = S3 + E * 128;    // [A,128]
  float* xe  = x + A * 128;     // [E,128]
  float* xf  = xe + E * 128;    // [F,128]
  float* cnt = xf + F * 128;    // [E] (covers A/F/B segment counts)
  float* st  = cnt + E;         // [256] bn stats
  float* pA  = st + 256;        // [B,128]
  float* pE  = pA + NB * 128;
  float* pF  = pE + NB * 128;
  size_t needed = (size_t)((pF + NB * 128) - S0) * sizeof(float);
  if (ws_size < needed) return;

  auto zero = [&](float* p, long long n) {
    hlg_zero<<<dim3((unsigned)((n + 255) / 256)), dim3(256), 0, stream>>>(p, n);
  };
  auto gemm = [&](const float* X0, const float* X1, const int* g0, const int* g1,
                  const float* W, const float* bias, float* Y, long long N, int K0, int K1, int relu) {
    hlg_gemm128<<<dim3((unsigned)((N + 127) / 128)), dim3(256), 0, stream>>>(
        X0, X1, g0, g1, W, bias, Y, N, K0, K1, relu);
  };
  auto segmean = [&](const float* src, const int* srcg, const int* idx, const int* idxg,
                     long long nsrc, long long nseg, float* dst) {
    zero(dst, nseg * 128);
    zero(cnt, nseg);
    hlg_scatter_add<<<dim3((unsigned)((nsrc * 128 + 255) / 256)), dim3(256), 0, stream>>>(src, srcg, idx, idxg, dst, nsrc);
    hlg_count<<<dim3((unsigned)((nsrc + 255) / 256)), dim3(256), 0, stream>>>(idx, idxg, cnt, nsrc);
    hlg_divide<<<dim3((unsigned)((nseg * 128 + 255) / 256)), dim3(256), 0, stream>>>(dst, cnt, nseg);
  };
  auto bnrelu = [&](const float* a, const float* b, float* y, long long n) {
    zero(st, 256);
    hlg_bn_stats<<<dim3((unsigned)((n + 255) / 256)), dim3(128), 0, stream>>>(a, b, st, n);
    hlg_bn_apply<<<dim3((unsigned)((n * 128 + 255) / 256)), dim3(256), 0, stream>>>(a, b, st, y, n);
  };

  // ---- encoders ----
  hlg_embed_atom<<<dim3((unsigned)((A * 128 + 255) / 256)), dim3(256), 0, stream>>>(x_atom, atom_emb, x, A);
  hlg_embed_bond<<<dim3((unsigned)((E * 128 + 255) / 256)), dim3(256), 0, stream>>>(edge_attr, bond_emb, xe, E);
  gemm(fragments, nullptr, nullptr, nullptr, frag_W, frag_b, xf, F, 32, 0, 0);

  // ---- layers ----
  for (int l = 0; l < 3; ++l) {
    auto Wa = [&](const float* base, int i) { return base + ((long long)l * 2 + i) * 128 * 128; };
    auto Ba = [&](const float* base, int i) { return base + ((long long)l * 2 + i) * 128; };

    // atom2atom: relu(concat(x[row_e], x_edge) @ Wb + bb) -> seg_mean(col_e) -> 2x(Lin+ReLU)
    gemm(x, xe, row_e, nullptr, a2a_Wb + (long long)l * 256 * 128, a2a_bb + l * 128, S0, E, 128, 128, 1);
    segmean(S0, nullptr, col_e, nullptr, E, A, S1);
    gemm(S1, nullptr, nullptr, nullptr, Wa(a2a_Wa, 0), Ba(a2a_ba, 0), S2, A, 128, 0, 1);
    gemm(S2, nullptr, nullptr, nullptr, Wa(a2a_Wa, 1), Ba(a2a_ba, 1), S1, A, 128, 0, 1);   // m_a2a
    // frag2atom: seg_mean(x_frag[ffi], fai) -> 2x(Lin+ReLU)
    segmean(xf, ffi, fai, nullptr, NAS, A, S2);
    gemm(S2, nullptr, nullptr, nullptr, Wa(f2a_Wa, 0), Ba(f2a_ba, 0), S3, A, 128, 0, 1);
    gemm(S3, nullptr, nullptr, nullptr, Wa(f2a_Wa, 1), Ba(f2a_ba, 1), S2, A, 128, 0, 1);   // m_f2a
    // combine + BN + residual BN
    gemm(S1, S2, nullptr, nullptr, cA_W + (long long)l * 256 * 128, cA_b + l * 128, S3, A, 128, 128, 0);
    bnrelu(S3, nullptr, S3, A);
    bnrelu(x, S3, x, A);

    // edge update (uses updated x)
    hlg_avg_gather<<<dim3((unsigned)((E * 128 + 255) / 256)), dim3(256), 0, stream>>>(x, row_e, col_e, S0, E);
    gemm(S0, nullptr, nullptr, nullptr, Wa(a2e_Wa, 0), Ba(a2e_ba, 0), S1, E, 128, 0, 1);
    gemm(S1, nullptr, nullptr, nullptr, Wa(a2e_Wa, 1), Ba(a2e_ba, 1), S0, E, 128, 0, 1);
    gemm(S0, nullptr, nullptr, nullptr, cE_W + (long long)l * 128 * 128, cE_b + l * 128, S1, E, 128, 0, 0);
    bnrelu(S1, nullptr, S1, E);
    bnrelu(xe, S1, xe, E);

    // fragment update (a2f uses updated x; f2f uses pre-update xf)
    segmean(x, fai, ffi, nullptr, NAS, F, S0);
    gemm(S0, nullptr, nullptr, nullptr, Wa(a2f_Wa, 0), Ba(a2f_ba, 0), S1, F, 128, 0, 1);
    gemm(S1, nullptr, nullptr, nullptr, Wa(a2f_Wa, 1), Ba(a2f_ba, 1), S0, F, 128, 0, 1);   // m_a2f
    segmean(xf, fr_row, fr_col, nullptr, FE, F, S1);
    gemm(S1, nullptr, nullptr, nullptr, Wa(f2f_Wa, 0), Ba(f2f_ba, 0), S2, F, 128, 0, 1);
    gemm(S2, nullptr, nullptr, nullptr, Wa(f2f_Wa, 1), Ba(f2f_ba, 1), S1, F, 128, 0, 1);   // m_f2f
    gemm(S0, S1, nullptr, nullptr, cF_W + (long long)l * 256 * 128, cF_b + l * 128, S2, F, 128, 128, 0);
    bnrelu(S2, nullptr, S2, F);
    bnrelu(xf, S2, xf, F);
  }

  // ---- readout ----
  // atoms
  gemm(x, nullptr, nullptr, nullptr, atom_out_W, atom_out_b, S0, A, 128, 0, 0);
  bnrelu(S0, nullptr, S0, A);
  gemm(S0, nullptr, nullptr, nullptr, atom_out_W + 128 * 128, atom_out_b + 128, S1, A, 128, 0, 0);
  bnrelu(S1, nullptr, S1, A);
  segmean(S1, nullptr, batch, nullptr, A, NB, pA);
  // edges (segment = batch[row_e])
  gemm(xe, nullptr, nullptr, nullptr, edge_out_W, edge_out_b, S0, E, 128, 0, 0);
  bnrelu(S0, nullptr, S0, E);
  gemm(S0, nullptr, nullptr, nullptr, edge_out_W + 128 * 128, edge_out_b + 128, S1, E, 128, 0, 0);
  bnrelu(S1, nullptr, S1, E);
  segmean(S1, nullptr, row_e, batch, E, NB, pE);
  // fragments
  gemm(xf, nullptr, nullptr, nullptr, frag_out_W, frag_out_b, S0, F, 128, 0, 0);
  bnrelu(S0, nullptr, S0, F);
  gemm(S0, nullptr, nullptr, nullptr, frag_out_W + 128 * 128, frag_out_b + 128, S1, F, 128, 0, 0);
  bnrelu(S1, nullptr, S1, F);
  segmean(S1, nullptr, frag_batch, nullptr, F, NB, pF);

  hlg_final<<<dim3(4), dim3(256), 0, stream>>>(pA, pE, pF, out_W, out_b, (float*)d_out);
}
